// CrossStageMoELayer_14972255994628
// MI455X (gfx1250) — matrix operand order repaired and checked by reading, not compiled, not run on hardware
//
#include <hip/hip_runtime.h>
#include <stdint.h>

// ---------------------------------------------------------------------------
// Problem constants (from reference): E=8 experts, C=1024, B=4, N=2048
// ---------------------------------------------------------------------------
constexpr int E_   = 8;
constexpr int C_   = 1024;
constexpr int B_   = 4;
constexpr int N_   = 2048;
constexpr int MTOT = 2 * N_;     // x rows ++ x_ir rows per batch = 4096
constexpr int GH   = 512;        // gate hidden dim = C/2

// GEMM tiling
constexpr int BM  = 128;
constexpr int BN  = 128;
constexpr int BK  = 64;
constexpr int LDT = BK + 8;      // padded LDS row stride in halves (144B: 16B aligned)

typedef _Float16 v16h __attribute__((ext_vector_type(16)));
typedef _Float16 h8   __attribute__((ext_vector_type(8)));
typedef _Float16 h4   __attribute__((ext_vector_type(4)));
typedef float    v8f  __attribute__((ext_vector_type(8)));

// Async global -> LDS copy, 16 bytes per lane. VDST is the wave-relative LDS
// byte address (low 32 bits of the generic pointer, per the flat-aperture
// truncation rule). Tracked by ASYNCcnt.
__device__ __forceinline__ void async_b128(uint32_t lds_addr, uint64_t gaddr) {
  asm volatile("global_load_async_to_lds_b128 %0, %1, off"
               :: "v"(lds_addr), "v"(gaddr)
               : "memory");
}
__device__ __forceinline__ void wait_async0() {
  asm volatile("s_wait_asynccnt 0" ::: "memory");
}

// ---------------------------------------------------------------------------
// 1) gx[b,c] = mean over n of x[b,n,c]
// ---------------------------------------------------------------------------
__global__ __launch_bounds__(256)
void mean_kernel(const float* __restrict__ x, float* __restrict__ gx) {
  int idx = blockIdx.x * 256 + threadIdx.x;        // 0..B*C-1 (4096)
  int b = idx >> 10;
  int c = idx & (C_ - 1);
  const float* p = x + ((size_t)b * N_) * C_ + c;
  float s = 0.f;
  for (int n = 0; n < N_; ++n) s += p[(size_t)n * C_];
  gx[idx] = s * (1.0f / (float)N_);
}

// ---------------------------------------------------------------------------
// 2) gate MLP + softmax -> g[B,E]   (single block)
// ---------------------------------------------------------------------------
__global__ __launch_bounds__(256)
void gate_kernel(const float* __restrict__ gx, const float* __restrict__ gw1,
                 const float* __restrict__ gb1, const float* __restrict__ gw2,
                 const float* __restrict__ gb2, float* __restrict__ g) {
  __shared__ float h[B_ * GH];
  __shared__ float lg[B_ * E_];
  const int tid = threadIdx.x;

  for (int idx = tid; idx < B_ * GH; idx += 256) {
    int b = idx >> 9;
    int j = idx & (GH - 1);
    const float* gr = gx  + (size_t)b * C_;
    const float* wr = gw1 + (size_t)j * C_;
    float s = gb1[j];
    for (int c = 0; c < C_; ++c) s += gr[c] * wr[c];
    h[idx] = s > 0.f ? s : 0.f;                    // ReLU
  }
  __syncthreads();

  if (tid < B_ * E_) {
    int b = tid >> 3;
    int e = tid & (E_ - 1);
    const float* hr = h   + b * GH;
    const float* wr = gw2 + e * GH;
    float s = gb2[e];
    for (int q = 0; q < GH; ++q) s += hr[q] * wr[q];
    lg[tid] = s;
  }
  __syncthreads();

  if (tid < B_) {
    float mx = -1e30f;
    for (int e = 0; e < E_; ++e) mx = fmaxf(mx, lg[tid * E_ + e]);
    float ex[E_], sum = 0.f;
    for (int e = 0; e < E_; ++e) { ex[e] = expf(lg[tid * E_ + e] - mx); sum += ex[e]; }
    float inv = 1.0f / sum;
    for (int e = 0; e < E_; ++e) g[tid * E_ + e] = ex[e] * inv;
  }
}

// ---------------------------------------------------------------------------
// 3) Wbh[b,o,c] = (f16) sum_e g[b,e]*We[e,o,c];  bb[b,o] = sum_e g[b,e]*be[e,o]
// ---------------------------------------------------------------------------
__global__ __launch_bounds__(256)
void wb_kernel(const float* __restrict__ We, const float* __restrict__ be,
               const float* __restrict__ g, _Float16* __restrict__ wbh,
               float* __restrict__ bb) {
  const int o = blockIdx.x;
  const int b = blockIdx.y;
  const int tid = threadIdx.x;
  __shared__ float gs[E_];
  if (tid < E_) gs[tid] = g[b * E_ + tid];
  __syncthreads();

  const int c0 = tid * 4;
  float ax = 0.f, ay = 0.f, az = 0.f, aw = 0.f;
  #pragma unroll
  for (int e = 0; e < E_; ++e) {
    const float4 w = *(const float4*)&We[((size_t)e * C_ + o) * C_ + c0];
    const float ge = gs[e];
    ax += ge * w.x; ay += ge * w.y; az += ge * w.z; aw += ge * w.w;
  }
  h4 outv = { (_Float16)ax, (_Float16)ay, (_Float16)az, (_Float16)aw };
  *(h4*)&wbh[((size_t)b * C_ + o) * C_ + c0] = outv;

  if (tid == 0) {
    float s = 0.f;
    #pragma unroll
    for (int e = 0; e < E_; ++e) s += gs[e] * be[(size_t)e * C_ + o];
    bb[b * C_ + o] = s;
  }
}

// ---------------------------------------------------------------------------
// 4) xh[b, m, c] (f16), m<2048 from x, m>=2048 from x_ir. 8 elems / thread.
// ---------------------------------------------------------------------------
__global__ __launch_bounds__(256)
void convert_kernel(const float* __restrict__ x, const float* __restrict__ xir,
                    _Float16* __restrict__ xh) {
  size_t base = ((size_t)blockIdx.x * 256 + threadIdx.x) * 8;   // flat into [B][MTOT][C]
  int c = (int)(base & (C_ - 1));
  int m = (int)((base >> 10) & (MTOT - 1));
  int b = (int)(base >> 22);
  const float* src = (m < N_) ? (x   + ((size_t)b * N_ + m)        * C_ + c)
                              : (xir + ((size_t)b * N_ + (m - N_)) * C_ + c);
  float4 f0 = *(const float4*)(src);
  float4 f1 = *(const float4*)(src + 4);
  h8 o;
  o[0] = (_Float16)f0.x; o[1] = (_Float16)f0.y; o[2] = (_Float16)f0.z; o[3] = (_Float16)f0.w;
  o[4] = (_Float16)f1.x; o[5] = (_Float16)f1.y; o[6] = (_Float16)f1.z; o[7] = (_Float16)f1.w;
  *(h8*)&xh[base] = o;
}

// ---------------------------------------------------------------------------
// 5) Main GEMM: out[m,o] = xh[b] (4096x1024) @ Wbh[b]^T + bb[b]
//    v_wmma_f32_16x16x32_f16; 128x128 tile, BK=64; double-buffered LDS fed by
//    global_load_async_to_lds_b128 (ASYNCcnt), one barrier per K step.
// ---------------------------------------------------------------------------
__global__ __launch_bounds__(256)
void moe_gemm_kernel(const _Float16* __restrict__ xh,    // [B][MTOT][C]
                     const _Float16* __restrict__ wbh,   // [B][O=C][C]
                     const float* __restrict__ bb,       // [B][C]
                     float* __restrict__ out) {          // moe_x ++ moe_x_ir
  __shared__ _Float16 As[2][BM * LDT];
  __shared__ _Float16 Bs[2][BN * LDT];

  const int m0 = blockIdx.x * BM;
  const int o0 = blockIdx.y * BN;
  const int b  = blockIdx.z;

  const int tid  = threadIdx.x;
  const int lane = tid & 31;
  const int wave = tid >> 5;
  const int wm0  = (wave & 1) * 64;     // 2 waves over M
  const int wn0  = (wave >> 1) * 32;    // 4 waves over N

  const _Float16* Ag = xh  + ((size_t)b * MTOT + m0) * C_;
  const _Float16* Bg = wbh + ((size_t)b * C_   + o0) * C_;

  // tile staging: each thread moves 4 rows x 8 halves (16B) per tile per matrix
  const int lr = tid >> 3;              // 0..31
  const int lc = (tid & 7) * 8;         // 0..56 (halves)

  uint64_t ga[4], gb[4];                // advance by BK*2 bytes per issued tile
  uint32_t la[4], lb[4];                // stage-0 LDS byte addresses
  #pragma unroll
  for (int i = 0; i < 4; ++i) {
    const int r = lr + i * 32;
    ga[i] = (uint64_t)(uintptr_t)(Ag + (size_t)r * C_ + lc);
    gb[i] = (uint64_t)(uintptr_t)(Bg + (size_t)r * C_ + lc);
    la[i] = (uint32_t)(uintptr_t)(&As[0][r * LDT + lc]);
    lb[i] = (uint32_t)(uintptr_t)(&Bs[0][r * LDT + lc]);
  }
  constexpr uint32_t STAGE_A = (uint32_t)(BM * LDT * sizeof(_Float16));  // As[1]-As[0]
  constexpr uint32_t STAGE_B = (uint32_t)(BN * LDT * sizeof(_Float16));

  // Issue one K-tile of async copies into stage s, then advance source ptrs.
  auto issue = [&](uint32_t soffA, uint32_t soffB) {
    #pragma unroll
    for (int i = 0; i < 4; ++i) {
      async_b128(la[i] + soffA, ga[i]);
      async_b128(lb[i] + soffB, gb[i]);
      ga[i] += (uint64_t)(BK * sizeof(_Float16));
      gb[i] += (uint64_t)(BK * sizeof(_Float16));
    }
  };

  v8f acc[4][2] = {};

  auto compute = [&](const _Float16* Asb, const _Float16* Bsb) {
    #pragma unroll
    for (int kk = 0; kk < BK; kk += 32) {
      // A frag: lane<16: M=lane, K=kk+[0..7] & kk+16+[0..7]; lane>=16 shift K by 8
      v16h af[4];
      const int ak = kk + ((lane & 16) ? 8 : 0);
      #pragma unroll
      for (int i = 0; i < 4; ++i) {
        const int m = wm0 + i * 16 + (lane & 15);
        const _Float16* p = &Asb[m * LDT + ak];
        v16h a;
        ((uint4*)&a)[0] = *(const uint4*)(p);
        ((uint4*)&a)[1] = *(const uint4*)(p + 16);
        af[i] = a;
      }
      // B frag: K=kk+[0..7] & kk+16+[0..7]; lane>=16 shift K by 8
      v16h bf[2];
      const int bk = kk + ((lane & 16) ? 8 : 0);
      #pragma unroll
      for (int j = 0; j < 2; ++j) {
        const int n = wn0 + j * 16 + (lane & 15);
        const _Float16* p = &Bsb[n * LDT + bk];
        v16h v;
        ((uint4*)&v)[0] = *(const uint4*)(p);
        ((uint4*)&v)[1] = *(const uint4*)(p + 16);
        bf[j] = v;
      }
      #pragma unroll
      for (int i = 0; i < 4; ++i)
        #pragma unroll
        for (int j = 0; j < 2; ++j)
          acc[i][j] = __builtin_amdgcn_wmma_f32_16x16x32_f16(
              false, af[i], false, bf[j], (short)0, acc[i][j], false, false);
    }
  };

  // prologue: fill stage 0
  issue(0, 0);
  wait_async0();
  __syncthreads();

  // C_/BK = 16 K-steps, processed as 8 double-steps with compile-time stages
  #pragma unroll 1
  for (int k0 = 0; k0 < C_; k0 += 2 * BK) {
    issue(STAGE_A, STAGE_B);            // fill stage 1 (k0+BK) while computing stage 0
    compute(As[0], Bs[0]);
    wait_async0();
    __syncthreads();

    if (k0 + 2 * BK < C_) issue(0, 0);  // fill stage 0 (k0+2BK) while computing stage 1
    compute(As[1], Bs[1]);
    wait_async0();
    __syncthreads();
  }

  // Epilogue: C layout — VGPR v: lanes 0-15 row v, lanes 16-31 row v+8; col = lane&15
  const int nlo   = lane & 15;
  const int mhalf = (lane & 16) ? 8 : 0;
  const size_t base_ir = (size_t)B_ * N_ * C_;
  #pragma unroll
  for (int j = 0; j < 2; ++j) {
    const int gn = o0 + wn0 + j * 16 + nlo;
    const float bias = bb[b * C_ + gn];
    #pragma unroll
    for (int i = 0; i < 4; ++i) {
      const int mbase = m0 + wm0 + i * 16 + mhalf;
      #pragma unroll
      for (int v = 0; v < 8; ++v) {
        const int gm = mbase + v;
        size_t off;
        if (gm < N_) off = ((size_t)b * N_ + gm) * C_ + gn;
        else         off = base_ir + ((size_t)b * N_ + (gm - N_)) * C_ + gn;
        out[off] = acc[i][j][v] + bias;
      }
    }
  }
}

// ---------------------------------------------------------------------------
// host launcher
// ---------------------------------------------------------------------------
extern "C" void kernel_launch(void* const* d_in, const int* in_sizes, int n_in,
                              void* d_out, int out_size, void* d_ws, size_t ws_size,
                              hipStream_t stream) {
  const float* x    = (const float*)d_in[0];
  const float* x_ir = (const float*)d_in[1];
  const float* We   = (const float*)d_in[2];
  const float* be   = (const float*)d_in[3];
  const float* gw1  = (const float*)d_in[4];
  const float* gb1  = (const float*)d_in[5];
  const float* gw2  = (const float*)d_in[6];
  const float* gb2  = (const float*)d_in[7];
  float* out = (float*)d_out;

  char* wsb = (char*)d_ws;
  _Float16* xh  = (_Float16*)(wsb);                         // 33,554,432 B
  _Float16* wbh = (_Float16*)(wsb + 33554432u);             //  8,388,608 B
  float*    bb  = (float*)(wsb + 41943040u);                //     16,384 B
  float*    gx  = (float*)(wsb + 41959424u);                //     16,384 B
  float*    g   = (float*)(wsb + 41975808u);                //        128 B

  mean_kernel<<<(B_ * C_) / 256, 256, 0, stream>>>(x, gx);
  gate_kernel<<<1, 256, 0, stream>>>(gx, gw1, gb1, gw2, gb2, g);
  wb_kernel<<<dim3(C_, B_), 256, 0, stream>>>(We, be, g, wbh, bb);
  {
    const size_t total = (size_t)B_ * MTOT * C_;            // 16,777,216 halves
    convert_kernel<<<(unsigned)(total / (256 * 8)), 256, 0, stream>>>(x, x_ir, xh);
  }
  moe_gemm_kernel<<<dim3(MTOT / BM, C_ / BN, B_), 256, 0, stream>>>(xh, wbh, bb, out);
}
